// ThreeLayerGCN_47227460387503
// MI455X (gfx1250) — compile-verified, matrix operand order checked by reading, not run on hardware
//
#include <hip/hip_runtime.h>

typedef __attribute__((ext_vector_type(16))) _Float16 v16h;
typedef __attribute__((ext_vector_type(8)))  float    v8f;

// ---------------------------------------------------------------------------
// Degree / normalization helpers
// ---------------------------------------------------------------------------
__global__ void k_init_deg(float* __restrict__ deg, int n) {
  int i = blockIdx.x * blockDim.x + threadIdx.x;
  if (i < n) deg[i] = 1.0f;  // self-loop contribution
}

__global__ void k_degree(const long long* __restrict__ dst, float* __restrict__ deg, int E) {
  int i = blockIdx.x * blockDim.x + threadIdx.x;
  if (i < E) atomicAdd(&deg[(int)dst[i]], 1.0f);
}

__global__ void k_rsqrt(const float* __restrict__ deg, float* __restrict__ dinv, int n) {
  int i = blockIdx.x * blockDim.x + threadIdx.x;
  if (i < n) {
    float d = deg[i];
    dinv[i] = (d > 0.0f) ? rsqrtf(d) : 0.0f;
  }
}

// ---------------------------------------------------------------------------
// Pre-pack W (f32 [128][dC]) into WMMA B-fragment layout, f16:
//   Wpk[frag][lane][e],  frag = ct*4 + kb,  lane in 0..31, e in 0..15
// so the GEMM loads one whole v16h fragment per (ct,kb,lane) with a single
// aligned 32-byte vector load. One thread per (frag,lane).
// ---------------------------------------------------------------------------
__global__ void k_pack_bfrag(const float* __restrict__ W, _Float16* __restrict__ Wpk,
                             int dC) {
  int t = blockIdx.x * blockDim.x + threadIdx.x;      // t = frag*32 + lane
  int nfrag = (dC / 16) * 4;                          // col tiles * 4 K-blocks
  if (t >= nfrag * 32) return;
  int lane  = t & 31;
  int frag  = t >> 5;
  int kb    = frag & 3;
  int ct    = frag >> 2;
  int n     = lane & 15;
  int khalf = lane >> 4;
  v16h vv;
#pragma unroll
  for (int e = 0; e < 16; ++e) {
    // element e -> K index (16-bit B-fragment layout, mirror of A, ISA 7.12.2)
    int k = kb * 32 + ((e < 8) ? (8 * khalf + e) : (16 + 8 * khalf + (e - 8)));
    vv[e] = (_Float16)W[(size_t)k * dC + ct * 16 + n];
  }
  *(v16h*)(Wpk + (size_t)t * 16) = vv;
}

// ---------------------------------------------------------------------------
// WMMA GEMM + dinv[row] epilogue:  H[m][c] = dinv[m] * sum_k A[m][k] * W[k][c]
// A: [M][128] f32 (converted to f16 at load via float4 vector loads),
// Wpk: pre-packed f16 B fragments, H: [M][DC] f32.  DC is compile-time so
// store strides become immediate offsets.
// Block = 256 threads = 8 waves; wave w owns 16-row strip (blockIdx.x*8+w)*16.
// K = 128 fixed -> 4x v_wmma_f32_16x16x32_f16 per 16x16 output tile.
// ---------------------------------------------------------------------------
template <int DC>
__global__ void k_gemm_wmma(const float* __restrict__ A,
                            const _Float16* __restrict__ Wpk,
                            const float* __restrict__ dinv,
                            float* __restrict__ H,
                            int M) {
  const int lane  = threadIdx.x & 31;
  const int wave  = threadIdx.x >> 5;
  const int r0    = (blockIdx.x * 8 + wave) * 16;
  if (r0 >= M) return;                 // wave-uniform: EXEC stays all-1s for WMMA

  const int m     = lane & 15;
  const int khalf = lane >> 4;         // 0: K 0..7 & 16..23 ; 1: K 8..15 & 24..31

  // ---- A fragments: 16 rows x K=128, f16, register-resident (4 x v16h) ----
  // Lane reads two contiguous 32B chunks per K-block -> 4x float4 loads.
  int rr = r0 + m; if (rr >= M) rr = M - 1;
  const float4* arow4 = (const float4*)(A + (size_t)rr * 128);
  v16h afrag[4];
#pragma unroll
  for (int kb = 0; kb < 4; ++kb) {
    int base = kb * 8 + khalf * 2;               // float4 units
    float4 c0 = arow4[base + 0];                 // k = kb*32 + 8*khalf + 0..3
    float4 c1 = arow4[base + 1];                 // k = ... + 4..7
    float4 c2 = arow4[base + 4];                 // k = kb*32 + 16 + 8*khalf + 0..3
    float4 c3 = arow4[base + 5];                 // k = ... + 4..7
    v16h a;
    a[0]  = (_Float16)c0.x; a[1]  = (_Float16)c0.y;
    a[2]  = (_Float16)c0.z; a[3]  = (_Float16)c0.w;
    a[4]  = (_Float16)c1.x; a[5]  = (_Float16)c1.y;
    a[6]  = (_Float16)c1.z; a[7]  = (_Float16)c1.w;
    a[8]  = (_Float16)c2.x; a[9]  = (_Float16)c2.y;
    a[10] = (_Float16)c2.z; a[11] = (_Float16)c2.w;
    a[12] = (_Float16)c3.x; a[13] = (_Float16)c3.y;
    a[14] = (_Float16)c3.z; a[15] = (_Float16)c3.w;
    afrag[kb] = a;
  }

  // dinv for the 8 output rows this lane-half writes (C layout: M = v + 8*khalf)
  float dv[8];
#pragma unroll
  for (int v = 0; v < 8; ++v) {
    int row = r0 + v + 8 * khalf; if (row >= M) row = M - 1;
    dv[v] = dinv[row];
  }

  const v16h* __restrict__ Wp = (const v16h*)Wpk;
  const int  n        = lane & 15;
  const bool fullTile = (r0 + 16 <= M);          // wave-uniform (always true here)

#pragma unroll
  for (int ct = 0; ct < DC / 16; ++ct) {
    // preload the 4 B fragments of this column tile (8x global_load_b128),
    // then chain the 4 WMMAs -> compiler can overlap loads with WMMA issue
    v16h bfrag[4];
#pragma unroll
    for (int kb = 0; kb < 4; ++kb)
      bfrag[kb] = Wp[(size_t)(ct * 4 + kb) * 32 + lane];

    v8f acc = {};
#pragma unroll
    for (int kb = 0; kb < 4; ++kb)
      acc = __builtin_amdgcn_wmma_f32_16x16x32_f16(
          /*neg_a=*/false, afrag[kb], /*neg_b=*/false, bfrag[kb],
          /*c_mod=*/(short)0, acc, /*reuse_a=*/false, /*reuse_b=*/false);

    // ---- store 16x16 tile, scaled by dinv[row] (C layout ISA 7.12.2) ----
    // Single base address; v*DC is a compile-time immediate offset.
    float* __restrict__ Hb = H + (size_t)(r0 + 8 * khalf) * DC + ct * 16 + n;
    if (fullTile) {                               // uniform branch: no EXEC churn
#pragma unroll
      for (int v = 0; v < 8; ++v) Hb[v * DC] = acc[v] * dv[v];
    } else {
#pragma unroll
      for (int v = 0; v < 8; ++v)
        if (r0 + v + 8 * khalf < M) Hb[v * DC] = acc[v] * dv[v];
    }
  }
}

// ---------------------------------------------------------------------------
// agg init = h' (the self-loop term), float4 copy
// ---------------------------------------------------------------------------
__global__ void k_copy4(const float* __restrict__ src, float* __restrict__ dst, int n4) {
  int i = blockIdx.x * blockDim.x + threadIdx.x;
  if (i < n4) ((float4*)dst)[i] = ((const float4*)src)[i];
}

// ---------------------------------------------------------------------------
// Edge scatter: one wave per edge. Lane reads D/32 contiguous floats of the
// source row (vector load), atomically accumulates into the dest row.
// Working set (h' + agg ~ 103 MB) is L2-resident on MI455X (192 MB L2).
// ---------------------------------------------------------------------------
template <int D>
__global__ void k_scatter(const long long* __restrict__ src,
                          const long long* __restrict__ dst,
                          const float* __restrict__ Hp,
                          float* __restrict__ agg, int E) {
  const int e    = (int)((blockIdx.x * (unsigned)blockDim.x + threadIdx.x) >> 5);
  const int lane = threadIdx.x & 31;
  if (e >= E) return;
  const int s = (int)src[e];          // broadcast load (all lanes same address)
  const int d = (int)dst[e];
  constexpr int FPL = D / 32;          // floats per lane (4 for D=128, 2 for D=64)
  const float* __restrict__ ps = Hp  + (size_t)s * D + lane * FPL;
  float*       __restrict__ pd = agg + (size_t)d * D + lane * FPL;
  if constexpr (FPL == 4) {
    float4 v = *(const float4*)ps;
    atomicAdd(pd + 0, v.x); atomicAdd(pd + 1, v.y);
    atomicAdd(pd + 2, v.z); atomicAdd(pd + 3, v.w);
  } else {
    float2 v = *(const float2*)ps;
    atomicAdd(pd + 0, v.x); atomicAdd(pd + 1, v.y);
  }
}

// ---------------------------------------------------------------------------
// Finalize: out = [relu]( agg * dinv[dst-row] + bias ), DC compile-time
// ---------------------------------------------------------------------------
template <int DC>
__global__ void k_finalize(const float* __restrict__ agg, const float* __restrict__ dinv,
                           const float* __restrict__ bias, float* __restrict__ out,
                           int total, int relu) {
  int i = blockIdx.x * blockDim.x + threadIdx.x;
  if (i >= total) return;
  int v = i / DC;          // DC is a power of two -> shift
  int c = i & (DC - 1);
  float o = agg[i] * dinv[v] + bias[c];
  if (relu) o = fmaxf(o, 0.0f);
  out[i] = o;
}

// ---------------------------------------------------------------------------
// Host orchestration
// ---------------------------------------------------------------------------
extern "C" void kernel_launch(void* const* d_in, const int* in_sizes, int n_in,
                              void* d_out, int out_size, void* d_ws, size_t ws_size,
                              hipStream_t stream) {
  const float*     x   = (const float*)d_in[0];
  const long long* ei  = (const long long*)d_in[1];   // int64 edge_index [2][E]
  const float*     W1  = (const float*)d_in[2];
  const float*     b1  = (const float*)d_in[3];
  const float*     W2  = (const float*)d_in[4];
  const float*     b2  = (const float*)d_in[5];
  const float*     W3  = (const float*)d_in[6];
  const float*     b3  = (const float*)d_in[7];
  float*           out = (float*)d_out;

  const int N = in_sizes[0] / 128;   // 100000 nodes
  const int E = in_sizes[1] / 2;     // 1600000 edges
  const long long* src = ei;
  const long long* dst = ei + E;

  // ---- carve workspace (deg, dinv, packed W, h', agg, inter-layer buf) ----
  size_t off = 0;
  auto carve = [&](size_t bytes) -> void* {
    void* p = (char*)d_ws + off;
    off += (bytes + 255) & ~(size_t)255;
    return p;
  };
  float*    deg  = (float*)carve((size_t)N * 4);
  float*    dinv = (float*)carve((size_t)N * 4);
  _Float16* Wpk  = (_Float16*)carve((size_t)128 * 128 * 2);   // packed B fragments
  float*    hpri = (float*)carve((size_t)N * 128 * 4);
  float*    agg  = (float*)carve((size_t)N * 128 * 4);
  float*    hbuf = (float*)carve((size_t)N * 128 * 4);
  (void)ws_size; (void)n_in; (void)out_size;

  // ---- symmetric normalization (shared by all 3 layers) ----
  k_init_deg<<<(N + 255) / 256, 256, 0, stream>>>(deg, N);
  k_degree<<<(E + 255) / 256, 256, 0, stream>>>(dst, deg, E);
  k_rsqrt<<<(N + 255) / 256, 256, 0, stream>>>(deg, dinv, N);

  const int rowTiles   = (N + 15) / 16;
  const int gemmBlocks = (rowTiles + 7) / 8;
  const int sblocks    = (E + 7) / 8;   // one wave per edge, 8 waves/block

  auto run_layer = [&](const float* in, const float* W, const float* b,
                       float* outp, int dC, bool relu) {
    // pack weights into f16 WMMA B-fragment layout (tiny, L2-resident)
    int packThreads = (dC / 16) * 4 * 32;
    k_pack_bfrag<<<(packThreads + 255) / 256, 256, 0, stream>>>(W, Wpk, dC);
    int total = N * dC;
    int n4 = total / 4;
    if (dC == 128) {
      k_gemm_wmma<128><<<gemmBlocks, 256, 0, stream>>>(in, Wpk, dinv, hpri, N);
      k_copy4<<<(n4 + 255) / 256, 256, 0, stream>>>(hpri, agg, n4);
      k_scatter<128><<<sblocks, 256, 0, stream>>>(src, dst, hpri, agg, E);
      k_finalize<128><<<(total + 255) / 256, 256, 0, stream>>>(agg, dinv, b, outp,
                                                               total, relu ? 1 : 0);
    } else {
      k_gemm_wmma<64><<<gemmBlocks, 256, 0, stream>>>(in, Wpk, dinv, hpri, N);
      k_copy4<<<(n4 + 255) / 256, 256, 0, stream>>>(hpri, agg, n4);
      k_scatter<64><<<sblocks, 256, 0, stream>>>(src, dst, hpri, agg, E);
      k_finalize<64><<<(total + 255) / 256, 256, 0, stream>>>(agg, dinv, b, outp,
                                                              total, relu ? 1 : 0);
    }
  };

  run_layer(x,    W1, b1, hbuf, 128, true);
  run_layer(hbuf, W2, b2, hbuf, 128, true);   // safe: stream-ordered, gemm consumes hbuf first
  run_layer(hbuf, W3, b3, out,  64,  false);
}
